// BidirDagLSTM_6038724018701
// MI455X (gfx1250) — matrix operand order repaired and checked by reading, not compiled, not run on hardware
//
#include <hip/hip_runtime.h>

// ---------------- static problem sizes (match reference) ----------------
#define L_LAYERS 8
#define NB       8192
#define EPN      2
#define EL       16384       // edges per layer
#define HDIM     256
#define NNODES   65536
#define NTYPE    9           // EC + 1
#define MAXTILES 1036        // >= EL/16 + NTYPE

typedef __attribute__((ext_vector_type(16))) __bf16          v16bf;
typedef __attribute__((ext_vector_type(8)))  float           v8f;
typedef __attribute__((ext_vector_type(8)))  unsigned short  ush8;

// fp32 -> bf16, round-to-nearest-even
__device__ inline unsigned short f2bf(float f) {
  unsigned int x = __float_as_uint(f);
  unsigned int r = x + 0x7fffu + ((x >> 16) & 1u);
  return (unsigned short)(r >> 16);
}

__device__ inline v16bf ld16(const unsigned short* p0, const unsigned short* p1) {
  union { unsigned short s[16]; v16bf v; } t;
  *reinterpret_cast<ush8*>(&t.s[0]) = *reinterpret_cast<const ush8*>(p0);
  *reinterpret_cast<ush8*>(&t.s[8]) = *reinterpret_cast<const ush8*>(p1);
  return t.v;
}

// A fragment (16x32 bf16) from LDS (16x256 row-major), per ISA 7.12.2 16-bit A layout.
__device__ inline v16bf load_a_frag(const unsigned short* aLds, int lane, int k0) {
  int row = lane & 15, h = lane >> 4;
  const unsigned short* p = aLds + row * 256 + k0 + 8 * h;
  return ld16(p, p + 16);
}

// B fragment (32x16) from n-major bf16 weight Bt[n][k] ([256][256]).
__device__ inline v16bf load_b_frag(const unsigned short* Bt, int lane, int n0, int k0) {
  int n = lane & 15, h = lane >> 4;
  const unsigned short* p = Bt + (size_t)(n0 + n) * 256 + k0 + 8 * h;
  return ld16(p, p + 16);
}

// C/D tile: elem r of lane -> row r + 8*(lane>>4), col n0 + (lane&15)
__device__ inline void store_acc(float* oLds, int lane, int n0, v8f acc) {
  int col = n0 + (lane & 15), h = lane >> 4;
#pragma unroll
  for (int r = 0; r < 8; ++r) oLds[(r + 8 * h) * 256 + col] = acc[r];
}

// stage 16 fp32 -> 16 bf16 into LDS as two 16-byte stores
__device__ inline void stage_bf16_row(unsigned short* dstLds, const float* src) {
  union { unsigned short s[16]; ush8 v[2]; } t;
#pragma unroll
  for (int i = 0; i < 16; ++i) t.s[i] = f2bf(src[i]);
  reinterpret_cast<ush8*>(dstLds)[0] = t.v[0];
  reinterpret_cast<ush8*>(dstLds)[1] = t.v[1];
}

struct Ln4  { const float* g[4]; const float* b[4]; };
struct CellP{ const float* bI; const float* bO; const float* bC; const float* bF;
              const float* lcg; const float* lcb; };

// ---------------- utility kernels ----------------
__global__ void zero_f32_kernel(float* p, int n) {
  int i = blockIdx.x * blockDim.x + threadIdx.x;
  if (i < n) p[i] = 0.f;
}
__global__ void zero_i32_kernel(int* p, int n) {
  int i = blockIdx.x * blockDim.x + threadIdx.x;
  if (i < n) p[i] = 0;
}

// Convert fp32 [mats][k][n] -> bf16 [mats][n][k] (transposed, n-major)
__global__ void convert_transpose_kernel(const float* __restrict__ src,
                                         unsigned short* __restrict__ dst, int mats) {
  int i = blockIdx.x * blockDim.x + threadIdx.x;
  int total = mats * 65536;
  if (i >= total) return;
  int m = i >> 16; int r = i & 65535; int k = r >> 8; int n = r & 255;
  dst[(size_t)m * 65536 + n * 256 + k] = f2bf(src[(size_t)m * 65536 + k * 256 + n]);
}

// ---------------- edge bucketing by (layer, type) ----------------
__global__ void bucket_count_kernel(const int* __restrict__ et, int* __restrict__ cnt) {
  int e = blockIdx.x * blockDim.x + threadIdx.x;
  if (e >= L_LAYERS * EL) return;
  int l = e / EL; int t = et[e];
  atomicAdd(&cnt[l * NTYPE + t], 1);
}

__global__ void bucket_build_kernel(const int* __restrict__ cnt, int* __restrict__ start,
                                    int4* __restrict__ descs, int* __restrict__ ntiles) {
  if (threadIdx.x != 0 || blockIdx.x != 0) return;
  for (int l = 0; l < L_LAYERS; ++l) {
    int s = 0, td = 0;
    for (int t = 0; t < NTYPE; ++t) {
      int c = cnt[l * NTYPE + t];
      start[l * NTYPE + t] = s;
      for (int off2 = 0; off2 < c; off2 += 16) {
        int nv = c - off2; if (nv > 16) nv = 16;
        descs[l * MAXTILES + td] = make_int4(s + off2, nv, t, 0);
        ++td;
      }
      s += c;
    }
    ntiles[l] = td;
  }
}

__global__ void bucket_scatter_kernel(const int* __restrict__ et, const int* __restrict__ start,
                                      int* __restrict__ cur, int* __restrict__ perm) {
  int e = blockIdx.x * blockDim.x + threadIdx.x;
  if (e >= L_LAYERS * EL) return;
  int l = e / EL, le = e - l * EL; int t = et[e];
  int pos = start[l * NTYPE + t] + atomicAdd(&cur[l * NTYPE + t], 1);
  perm[l * EL + pos] = le;
}

// ---------------- shared GEMM body: fused-4-gate 16x(4*256)x256 WMMA + per-row LN ----
// A (16x256 bf16) is in aLds; Bg[g] point to n-major [256][256] bf16 weights.
// Single K loop keeps 8 accumulators (4 gates x 2 N-tiles) live so every A
// fragment is read from LDS exactly once; LN+store runs per gate reusing oLds.
struct GemmOut {
  float* oLds;                    // 16*256 staging
  float (*redS)[16]; float (*redQ)[16];
  float* rowM; float* rowR;
};

template <typename WriteFn>
__device__ inline void gemm4_ln(const unsigned short* aLds,
                                const unsigned short* Bg0, const unsigned short* Bg1,
                                const unsigned short* Bg2, const unsigned short* Bg3,
                                GemmOut go, Ln4 ln, int tid, WriteFn writeRow) {
  const unsigned short* Bg[4] = {Bg0, Bg1, Bg2, Bg3};
  int lane = tid & 31, w = tid >> 5;
  int n0 = w * 32;
  v8f acc[4][2];
#pragma unroll
  for (int g = 0; g < 4; ++g) { acc[g][0] = {}; acc[g][1] = {}; }
#pragma unroll
  for (int k0 = 0; k0 < 256; k0 += 32) {
    v16bf a = load_a_frag(aLds, lane, k0);
#pragma unroll
    for (int g = 0; g < 4; ++g) {
      v16bf b0 = load_b_frag(Bg[g], lane, n0,      k0);
      v16bf b1 = load_b_frag(Bg[g], lane, n0 + 16, k0);
      acc[g][0] = __builtin_amdgcn_wmma_f32_16x16x32_bf16(false, a, false, b0, (short)0, acc[g][0], false, false);
      acc[g][1] = __builtin_amdgcn_wmma_f32_16x16x32_bf16(false, a, false, b1, (short)0, acc[g][1], false, false);
    }
  }
  // per-gate: stage to LDS, row LayerNorm over 256 cols, write out
  for (int g = 0; g < 4; ++g) {
    store_acc(go.oLds, lane, n0,      acc[g][0]);
    store_acc(go.oLds, lane, n0 + 16, acc[g][1]);
    __syncthreads();
    int r = tid >> 4, seg = tid & 15;
    const float* o = go.oLds + r * 256 + seg * 16;
    float s = 0.f, q = 0.f;
#pragma unroll
    for (int i = 0; i < 16; ++i) { float v = o[i]; s += v; q += v * v; }
    go.redS[r][seg] = s; go.redQ[r][seg] = q;
    __syncthreads();
    if (tid < 16) {
      float ss = 0.f, qq = 0.f;
#pragma unroll
      for (int i = 0; i < 16; ++i) { ss += go.redS[tid][i]; qq += go.redQ[tid][i]; }
      float m = ss * (1.f / 256.f);
      float var = qq * (1.f / 256.f) - m * m;
      go.rowM[tid] = m; go.rowR[tid] = rsqrtf(var + 1e-5f);
    }
    __syncthreads();
    writeRow(g, r, seg, o, go.rowM[r], go.rowR[r], ln.g[g] + seg * 16, ln.b[g] + seg * 16);
    __syncthreads();
  }
}

// ---------------- Ww projection: LN(x @ W_g) for this layer's 8192 nodes ----------------
__global__ __launch_bounds__(256)
void ww_proj_kernel(int nodeBase, const float* __restrict__ x,
                    const unsigned short* __restrict__ Wt /* [4][256][256] n-major bf16 */,
                    Ln4 ln, float* __restrict__ ww /* [NB][1024] */) {
  __shared__ unsigned short aLds[16 * 256];
  __shared__ float oLds[16 * 256];
  __shared__ float redS[16][16], redQ[16][16];
  __shared__ float rowM[16], rowR[16];
  int tid = threadIdx.x;
  int row0 = blockIdx.x * 16;
  {
    int r = tid >> 4, cb = (tid & 15) * 16;
    stage_bf16_row(aLds + r * 256 + cb, x + (size_t)(nodeBase + row0 + r) * 256 + cb);
  }
  __syncthreads();
  GemmOut go{oLds, redS, redQ, rowM, rowR};
  gemm4_ln(aLds, Wt, Wt + 65536, Wt + 2 * 65536, Wt + 3 * 65536, go, ln, tid,
    [&](int g, int r, int seg, const float* o, float m, float rs,
        const float* lg, const float* lb) {
      float* dst = ww + (size_t)(row0 + r) * 1024 + g * 256 + seg * 16;
#pragma unroll
      for (int i = 0; i < 16; ++i) dst[i] = lg[i] * (o[i] - m) * rs + lb[i];
    });
}

// ---------------- edge gates: LN(reprs[a] @ U_g[type]) per edge ----------------
__global__ __launch_bounds__(256)
void edge_gate_kernel(int l, const int4* __restrict__ descs, const int* __restrict__ ntiles,
                      const int* __restrict__ perm, const int* __restrict__ add,
                      const float* __restrict__ reprs,
                      const unsigned short* __restrict__ Ut /* [4][9][256][256] n-major bf16 */,
                      Ln4 ln, float* __restrict__ uBuf /* [EL][1024] */) {
  int nt = ntiles[l];
  if ((int)blockIdx.x >= nt) return;
  __shared__ unsigned short aLds[16 * 256];
  __shared__ float oLds[16 * 256];
  __shared__ float redS[16][16], redQ[16][16];
  __shared__ float rowM[16], rowR[16];
  __shared__ int eIdx[16], aIdx[16], tType[1];
  int tid = threadIdx.x;
  if (tid < 16) {
    int4 dsc = descs[l * MAXTILES + blockIdx.x];
    if (tid == 0) tType[0] = dsc.z;
    int e = (tid < dsc.y) ? perm[l * EL + dsc.x + tid] : -1;
    eIdx[tid] = e;
    aIdx[tid] = (e >= 0) ? add[l * EL + e] : NNODES;  // row NNODES stays zero
  }
  __syncthreads();
  {
    int r = tid >> 4, cb = (tid & 15) * 16;
    stage_bf16_row(aLds + r * 256 + cb, reprs + (size_t)aIdx[r] * 256 + cb);
  }
  __syncthreads();
  int t = tType[0];
  const unsigned short* B0 = Ut + (size_t)(0 * NTYPE + t) * 65536;
  const unsigned short* B1 = Ut + (size_t)(1 * NTYPE + t) * 65536;
  const unsigned short* B2 = Ut + (size_t)(2 * NTYPE + t) * 65536;
  const unsigned short* B3 = Ut + (size_t)(3 * NTYPE + t) * 65536;
  GemmOut go{oLds, redS, redQ, rowM, rowR};
  gemm4_ln(aLds, B0, B1, B2, B3, go, ln, tid,
    [&](int g, int r, int seg, const float* o, float m, float rs,
        const float* lg, const float* lb) {
      int e = eIdx[r];
      if (e >= 0) {
        float* dst = uBuf + (size_t)e * 1024 + g * 256 + seg * 16;
#pragma unroll
        for (int i = 0; i < 16; ++i) dst[i] = lg[i] * (o[i] - m) * rs + lb[i];
      }
    });
}

// ---------------- node update: gates + cell LN + output ----------------
__global__ __launch_bounds__(256)
void node_update_kernel(int l, int nodeBase,
                        const float* __restrict__ ww, const float* __restrict__ uBuf,
                        const int* __restrict__ add, float* __restrict__ memv,
                        float* __restrict__ reprs, CellP cp,
                        float* __restrict__ out, int dirCol) {
  __shared__ float s1[256], s2[256];
  int k = blockIdx.x;                 // node within layer (edges 2k, 2k+1)
  int j = nodeBase + k;               // global node id
  int h = threadIdx.x;
  int e0 = 2 * k, e1 = 2 * k + 1;
  int a0 = add[l * EL + e0], a1 = add[l * EL + e1];
  const float* wwr = ww   + (size_t)k  * 1024;
  const float* u0  = uBuf + (size_t)e0 * 1024;
  const float* u1  = uBuf + (size_t)e1 * 1024;
  float bI = *cp.bI, bO = *cp.bO, bC = *cp.bC, bF = *cp.bF;
  float iG = 1.f / (1.f + expf(-(wwr[0 * 256 + h] + u0[0 * 256 + h] + u1[0 * 256 + h] + bI)));
  float oG = 1.f / (1.f + expf(-(wwr[1 * 256 + h] + u0[1 * 256 + h] + u1[1 * 256 + h] + bO)));
  float cG = tanhf(wwr[2 * 256 + h] + u0[2 * 256 + h] + u1[2 * 256 + h] + bC);
  float wf = wwr[3 * 256 + h];
  float f0 = 1.f / (1.f + expf(-(wf + u0[3 * 256 + h] + bF)));
  float f1 = 1.f / (1.f + expf(-(wf + u1[3 * 256 + h] + bF)));
  float m0 = memv[(size_t)a0 * 256 + h], m1 = memv[(size_t)a1 * 256 + h];
  float par = iG * cG + f0 * m0 + f1 * m1;
  memv[(size_t)j * 256 + h] = par;
  s1[h] = par; s2[h] = par * par;
  __syncthreads();
  for (int o = 128; o > 0; o >>= 1) {
    if (h < o) { s1[h] += s1[h + o]; s2[h] += s2[h + o]; }
    __syncthreads();
  }
  float m = s1[0] * (1.f / 256.f);
  float var = s2[0] * (1.f / 256.f) - m * m;
  float lnv = cp.lcg[h] * (par - m) * rsqrtf(var + 1e-5f) + cp.lcb[h];
  float o2 = oG * tanhf(lnv);
  reprs[(size_t)j * 256 + h] = o2;
  out[(size_t)j * 512 + dirCol * 256 + h] = o2;
}

// ---------------- host launch ----------------
extern "C" void kernel_launch(void* const* d_in, const int* in_sizes, int n_in,
                              void* d_out, int out_size, void* d_ws, size_t ws_size,
                              hipStream_t stream) {
  (void)in_sizes; (void)n_in; (void)out_size; (void)ws_size;
  const float* node_states = (const float*)d_in[0];

  char* ws = (char*)d_ws;
  size_t off = 0;
  auto carve = [&](size_t bytes) -> char* {
    char* p = ws + off; off += (bytes + 255) & ~(size_t)255; return p;
  };
  unsigned short* Wt = (unsigned short*)carve((size_t)2 * 4 * 65536 * 2);
  unsigned short* Ut = (unsigned short*)carve((size_t)2 * 4 * 9 * 65536 * 2);
  float* WwBuf = (float*)carve((size_t)NB * 1024 * 4);
  float* uBuf  = (float*)carve((size_t)EL * 1024 * 4);
  float* reprs = (float*)carve((size_t)(NNODES + 1) * 256 * 4);
  float* memb  = (float*)carve((size_t)(NNODES + 1) * 256 * 4);
  int*   perm  = (int*)carve((size_t)L_LAYERS * EL * 4);
  int*   cnt   = (int*)carve(L_LAYERS * NTYPE * 4);
  int*   start = (int*)carve(L_LAYERS * NTYPE * 4);
  int*   cur   = (int*)carve(L_LAYERS * NTYPE * 4);
  int*   ntile = (int*)carve(L_LAYERS * 4);
  int4*  descs = (int4*)carve((size_t)L_LAYERS * MAXTILES * 16);

  // params: jax-tree (sorted) flattening -> dn leaves at d_in[1..30], up at d_in[31..60]
  const int gU[4] = {2, 3, 0, 1};   // internal gate order i,o,c,f -> U_c,U_f,U_i,U_o offsets
  for (int dir = 0; dir < 2; ++dir) {          // dir 0 = up, 1 = dn
    int B = (dir == 0) ? 31 : 1;
    for (int g = 0; g < 4; ++g) {
      const float* Wsrc = (const float*)d_in[B + 4 + gU[g]];
      const float* Usrc = (const float*)d_in[B + 0 + gU[g]];
      unsigned short* Wdst = Wt + ((size_t)dir * 4 + g) * 65536;
      unsigned short* Udst = Ut + ((size_t)dir * 4 + g) * 9 * 65536;
      convert_transpose_kernel<<<(1 * 65536) / 256, 256, 0, stream>>>(Wsrc, Wdst, 1);
      convert_transpose_kernel<<<(9 * 65536) / 256, 256, 0, stream>>>(Usrc, Udst, 9);
    }
  }

  float* outp = (float*)d_out;
  for (int dir = 0; dir < 2; ++dir) {
    int B = (dir == 0) ? 31 : 1;
    const int* add = (const int*)d_in[(dir == 0) ? 62 : 65];
    const int* et  = (const int*)d_in[(dir == 0) ? 63 : 66];

    int nz = (NNODES + 1) * 256;
    zero_f32_kernel<<<(nz + 255) / 256, 256, 0, stream>>>(reprs, nz);
    zero_f32_kernel<<<(nz + 255) / 256, 256, 0, stream>>>(memb, nz);
    zero_i32_kernel<<<1, 256, 0, stream>>>(cnt, L_LAYERS * NTYPE);
    zero_i32_kernel<<<1, 256, 0, stream>>>(cur, L_LAYERS * NTYPE);
    bucket_count_kernel<<<(L_LAYERS * EL) / 256, 256, 0, stream>>>(et, cnt);
    bucket_build_kernel<<<1, 32, 0, stream>>>(cnt, start, descs, ntile);
    bucket_scatter_kernel<<<(L_LAYERS * EL) / 256, 256, 0, stream>>>(et, start, cur, perm);

    Ln4 lnw, lnu;
    const int lwg[4] = {27, 29, 23, 25}, lwb[4] = {26, 28, 22, 24};
    const int lug[4] = {19, 21, 15, 17}, lub[4] = {18, 20, 14, 16};
    for (int g = 0; g < 4; ++g) {
      lnw.g[g] = (const float*)d_in[B + lwg[g]]; lnw.b[g] = (const float*)d_in[B + lwb[g]];
      lnu.g[g] = (const float*)d_in[B + lug[g]]; lnu.b[g] = (const float*)d_in[B + lub[g]];
    }
    CellP cp;
    cp.bI = (const float*)d_in[B + 10]; cp.bO = (const float*)d_in[B + 11];
    cp.bC = (const float*)d_in[B + 8];  cp.bF = (const float*)d_in[B + 9];
    cp.lcg = (const float*)d_in[B + 13]; cp.lcb = (const float*)d_in[B + 12];

    const unsigned short* WtD = Wt + (size_t)dir * 4 * 65536;
    const unsigned short* UtD = Ut + (size_t)dir * 4 * 9 * 65536;

    for (int l = 0; l < L_LAYERS; ++l) {
      int nodeBase = ((dir == 0) ? l : (L_LAYERS - 1 - l)) * NB;
      ww_proj_kernel<<<NB / 16, 256, 0, stream>>>(nodeBase, node_states, WtD, lnw, WwBuf);
      edge_gate_kernel<<<MAXTILES, 256, 0, stream>>>(l, descs, ntile, perm, add, reprs, UtD, lnu, uBuf);
      node_update_kernel<<<NB, 256, 0, stream>>>(l, nodeBase, WwBuf, uBuf, add, memb, reprs, cp, outp, dir);
    }
  }
}
